// DiffPool_reg_64278480552408
// MI455X (gfx1250) — compile-verified, matrix operand order checked by reading, not run on hardware
//
#include <hip/hip_runtime.h>

#define NNODE 64
#define CIN   128
#define HIDC  128
#define KCL   5
#define DEG   8
#define EPG   (NNODE * DEG)   // 512 edges per graph

typedef __attribute__((ext_vector_type(16))) _Float16 v16h;
typedef __attribute__((ext_vector_type(8)))  _Float16 v8h;
typedef __attribute__((ext_vector_type(8)))  float    v8f;

// ---- WMMA fragment loaders (CDNA5 wave32 16x16x32 f16 layouts) ----
// A (16x32, MxK): lane -> m = lane&15, half = lane>>4;
//   element e -> k = 16*(e>>3) + 8*half + (e&7)   (two contiguous runs of 8)
__device__ __forceinline__ v16h frag_a_lds(const _Float16* base, int ld,
                                           int row0, int k0, int lane) {
    int m = lane & 15, hf = lane >> 4;
    const _Float16* p = base + (row0 + m) * ld + k0 + 8 * hf;
    v8h lo = *(const v8h*)p;
    v8h hi = *(const v8h*)(p + 16);
    v16h f;
#pragma unroll
    for (int e = 0; e < 8; ++e) { f[e] = lo[e]; f[e + 8] = hi[e]; }
    return f;
}

// B (32x16, KxN) taken from a TRANSPOSED LDS buffer baseT[n][k] (ld = k-dim)
// so each lane's 16 k-values are two contiguous 16B loads.
__device__ __forceinline__ v16h frag_b_ldsT(const _Float16* baseT, int ld,
                                            int k0, int c0, int lane) {
    int n = lane & 15, hf = lane >> 4;
    const _Float16* p = baseT + (c0 + n) * ld + k0 + 8 * hf;
    v8h lo = *(const v8h*)p;
    v8h hi = *(const v8h*)(p + 16);
    v16h f;
#pragma unroll
    for (int e = 0; e < 8; ++e) { f[e] = lo[e]; f[e + 8] = hi[e]; }
    return f;
}

// B (32x16) from row-major global f32 weight W[k][c]; lanes sharing a k-row
// read 16 consecutive floats -> coalesced; W1 stays resident in L2.
__device__ __forceinline__ v16h frag_b_gf32(const float* __restrict__ W, int ld,
                                            int k0, int c0, int lane) {
    int n = lane & 15, hf = lane >> 4;
    v16h f;
#pragma unroll
    for (int e = 0; e < 16; ++e) {
        int k = ((e >> 3) << 4) + 8 * hf + (e & 7);
        f[e] = (_Float16)W[(k0 + k) * ld + c0 + n];
    }
    return f;
}

__global__ __launch_bounds__(256)
void diffpool_kernel(const float* __restrict__ x,
                     const int*   __restrict__ ei,   // [2, E] int32
                     const float* __restrict__ Wp, const float* __restrict__ bp,
                     const float* __restrict__ W1, const float* __restrict__ b1,
                     const float* __restrict__ W2, const float* __restrict__ b2,
                     const float* __restrict__ Wl, const float* __restrict__ bl,
                     float* __restrict__ out, int E)
{
    __shared__ _Float16 s_xh[NNODE * CIN];      // 16KB: x (f16); later x_l1
    __shared__ float    s_adj[NNODE * NNODE];   // 16KB: adj f32; later tail scratch
    __shared__ _Float16 s_adjn[NNODE * NNODE];  //  8KB: normalized adj (f16)
    __shared__ _Float16 s_yT[HIDC * NNODE];     // 16KB: (x@W1)^T  [c][j]
    __shared__ float    s_s[NNODE * 8];         //  2KB: s_pre -> softmax s (K pad 8)
    __shared__ float    s_v[NNODE * 8];         //  2KB: xWp, then t = adj@s
    __shared__ float    s_d[NNODE];

    const int tid  = threadIdx.x;
    const int lane = tid & 31;
    const int wave = tid >> 5;
    const int g    = blockIdx.x;

    // ---- phase 0: zero adj, stage x as f16 ----
    for (int i = tid; i < NNODE * NNODE; i += 256) s_adj[i] = 0.0f;
    const float* xg = x + (size_t)g * NNODE * CIN;
    for (int i = tid; i < NNODE * CIN; i += 256) s_xh[i] = (_Float16)xg[i];
    __syncthreads();

    // ---- phase 1: scatter edges (contiguous block of 512 per graph) ----
    {
        const int e0 = g * EPG;
        const int nbase = g * NNODE;
        for (int e = tid; e < EPG; e += 256) {
            int idx = e0 + e;
            int r = ei[idx] - nbase;          // src pos in graph
            int c = ei[E + idx] - nbase;      // dst pos in graph
            atomicAdd(&s_adj[r * NNODE + c], 1.0f);
        }
    }
    __syncthreads();

    // ---- phase 2: degree (with forced unit diagonal) and normalized adj ----
    if (tid < NNODE) {
        float sum = 0.0f;
        for (int j = 0; j < NNODE; ++j)
            sum += (j == tid) ? 1.0f : s_adj[tid * NNODE + j];
        s_d[tid] = rsqrtf(fmaxf(sum, 1.0f));
    }
    __syncthreads();
    for (int i = tid; i < NNODE * NNODE; i += 256) {
        int r = i >> 6, c = i & 63;
        float a = (r == c) ? 1.0f : s_adj[i];
        s_adjn[i] = (_Float16)(s_d[r] * a * s_d[c]);
    }
    __syncthreads();

    // ---- phase 3: WMMA GEMM1: xW1 = x @ W1 -> s_yT (transposed, f16) ----
    {
        const int c0 = wave * 16;               // 8 waves cover 128 cols
        v8f acc[4] = {};
#pragma unroll
        for (int kc = 0; kc < 4; ++kc) {        // K = 128 = 4 x 32
            v16h bf = frag_b_gf32(W1, HIDC, kc * 32, c0, lane);
#pragma unroll
            for (int mt = 0; mt < 4; ++mt) {    // 64 rows = 4 x 16
                v16h af = frag_a_lds(s_xh, CIN, mt * 16, kc * 32, lane);
                acc[mt] = __builtin_amdgcn_wmma_f32_16x16x32_f16(
                    false, af, false, bf, (short)0, acc[mt], false, false);
            }
        }
        int n = lane & 15, hf = lane >> 4;
#pragma unroll
        for (int mt = 0; mt < 4; ++mt) {
            v8h st;
#pragma unroll
            for (int r = 0; r < 8; ++r) st[r] = (_Float16)acc[mt][r];
            *(v8h*)&s_yT[(c0 + n) * NNODE + mt * 16 + 8 * hf] = st;
        }
    }
    // VALU (same phase, only reads s_xh): xWp = x @ W_pool  [64,5]
    for (int o = tid; o < NNODE * KCL; o += 256) {
        int i = o / KCL, j = o % KCL;
        float sum = 0.0f;
        for (int k = 0; k < CIN; ++k)
            sum += (float)s_xh[i * CIN + k] * Wp[k * KCL + j];
        s_v[i * 8 + j] = sum;
    }
    __syncthreads();

    // ---- phase 4: s_pre = adj_n @ xWp + b_pool ----
    for (int o = tid; o < NNODE * KCL; o += 256) {
        int i = o / KCL, j = o % KCL;
        float sum = bp[j];
        for (int m = 0; m < NNODE; ++m)
            sum += (float)s_adjn[i * NNODE + m] * s_v[m * 8 + j];
        s_s[i * 8 + j] = sum;
    }
    __syncthreads();

    // ---- phase 5: row softmax over K=5 ----
    if (tid < NNODE) {
        float mx = -1e30f;
        for (int j = 0; j < KCL; ++j) mx = fmaxf(mx, s_s[tid * 8 + j]);
        float e[KCL], sum = 0.0f;
        for (int j = 0; j < KCL; ++j) { e[j] = __expf(s_s[tid * 8 + j] - mx); sum += e[j]; }
        float inv = 1.0f / sum;
        for (int j = 0; j < KCL; ++j) s_s[tid * 8 + j] = e[j] * inv;
    }
    __syncthreads();

    // ---- phase 6: WMMA GEMM2: x_l1 = adj_n @ xW1 + b1 -> s_xh (f16) ----
    {
        const int c0 = wave * 16;
        v8f acc[4] = {};
#pragma unroll
        for (int kc = 0; kc < 2; ++kc) {        // K = 64 = 2 x 32
            v16h bf = frag_b_ldsT(s_yT, NNODE, kc * 32, c0, lane);
#pragma unroll
            for (int mt = 0; mt < 4; ++mt) {
                v16h af = frag_a_lds(s_adjn, NNODE, mt * 16, kc * 32, lane);
                acc[mt] = __builtin_amdgcn_wmma_f32_16x16x32_f16(
                    false, af, false, bf, (short)0, acc[mt], false, false);
            }
        }
        int n = lane & 15, hf = lane >> 4;
        float bias = b1[c0 + n];
#pragma unroll
        for (int mt = 0; mt < 4; ++mt)
#pragma unroll
            for (int r = 0; r < 8; ++r) {
                int m = mt * 16 + 8 * hf + r;
                s_xh[m * CIN + c0 + n] = (_Float16)(acc[mt][r] + bias);
            }
    }
    __syncthreads();

    // ---- phase 7: t = adj @ s  [64,5] (last reader of f32 adj) ----
    for (int o = tid; o < NNODE * KCL; o += 256) {
        int i = o / KCL, j = o % KCL;
        float sum = 0.0f;
        for (int m = 0; m < NNODE; ++m)
            sum += s_adj[i * NNODE + m] * s_s[m * 8 + j];
        s_v[i * 8 + j] = sum;
    }
    __syncthreads();

    // adj region is dead -> alias tail scratch into it
    float* s_xp1 = s_adj;           // [5][128]
    float* s_y2  = s_adj + 640;     // [5][128]
    float* s_ap  = s_adj + 1280;    // [5][5]
    float* s_red = s_adj + 1312;    // [128]

    // ---- phase 8: adj_p1 = s^T t ; x_p1 = s^T x_l1 ----
    for (int o = tid; o < KCL * KCL; o += 256) {
        int k = o / KCL, l = o % KCL;
        float sum = 0.0f;
        for (int m = 0; m < NNODE; ++m)
            sum += s_s[m * 8 + k] * s_v[m * 8 + l];
        s_ap[o] = sum;
    }
    for (int o = tid; o < KCL * HIDC; o += 256) {
        int k = o / HIDC, c = o % HIDC;
        float sum = 0.0f;
        for (int m = 0; m < NNODE; ++m)
            sum += s_s[m * 8 + k] * (float)s_xh[m * CIN + c];
        s_xp1[o] = sum;
    }
    __syncthreads();

    // ---- phase 9: y2 = x_p1 @ W2  [5,128] (coalesced over c) ----
    for (int o = tid; o < KCL * HIDC; o += 256) {
        int k = o / HIDC, c = o % HIDC;
        float sum = 0.0f;
        for (int h = 0; h < HIDC; ++h)
            sum += s_xp1[k * HIDC + h] * W2[h * HIDC + c];
        s_y2[o] = sum;
    }
    __syncthreads();

    // ---- phase 10: x_l2 = adj_n2 @ y2 + b2 ; pooled ; readout ----
    if (tid < HIDC) {
        float d2[KCL];
#pragma unroll
        for (int k = 0; k < KCL; ++k) {
            float sum = 0.0f;
            for (int l = 0; l < KCL; ++l)
                sum += (k == l) ? 1.0f : s_ap[k * KCL + l];
            d2[k] = rsqrtf(fmaxf(sum, 1.0f));
        }
        float pooled = 0.0f;
#pragma unroll
        for (int k = 0; k < KCL; ++k) {
            float v = b2[tid];
            for (int l = 0; l < KCL; ++l) {
                float a = (k == l) ? 1.0f : s_ap[k * KCL + l];
                v += d2[k] * a * d2[l] * s_y2[l * HIDC + tid];
            }
            pooled += v;
        }
        s_red[tid] = pooled * Wl[tid];
    }
    __syncthreads();
    if (tid == 0) {
        float sum = bl[0];
        for (int c = 0; c < HIDC; ++c) sum += s_red[c];
        out[g] = sum;
    }
}

extern "C" void kernel_launch(void* const* d_in, const int* in_sizes, int n_in,
                              void* d_out, int out_size, void* d_ws, size_t ws_size,
                              hipStream_t stream) {
    // setup_inputs order: x, edge_index, batch, W_pool, b_pool, W1, b1,
    //                     W2, b2, W_lin, b_lin, num_graphs, max_nodes
    const float* x  = (const float*)d_in[0];
    const int*   ei = (const int*)  d_in[1];
    const float* Wp = (const float*)d_in[3];
    const float* bp = (const float*)d_in[4];
    const float* W1 = (const float*)d_in[5];
    const float* b1 = (const float*)d_in[6];
    const float* W2 = (const float*)d_in[7];
    const float* b2 = (const float*)d_in[8];
    const float* Wl = (const float*)d_in[9];
    const float* bl = (const float*)d_in[10];
    const int E = in_sizes[1] / 2;
    const int B = out_size;                 // one scalar per graph

    hipLaunchKernelGGL(diffpool_kernel, dim3(B), dim3(256), 0, stream,
                       x, ei, Wp, bp, W1, b1, W2, b2, Wl, bl,
                       (float*)d_out, E);
}
